// sLSTMBlock_8985071583365
// MI455X (gfx1250) — matrix operand order repaired and checked by reading, not compiled, not run on hardware
//
#include <hip/hip_runtime.h>
#include <hip/hip_bf16.h>
#include <math.h>

// ---------------------------------------------------------------------------
// sLSTM block forward for MI455X (gfx1250, wave32, WMMA bf16 path)
//   B=4096, D=HID=2048, H=8 heads, HS=256, P=2730
// Weights are pre-swizzled into WMMA B-fragment order at staging time so the
// GEMM inner loop is pure b128 loads + v_wmma.
// ---------------------------------------------------------------------------

typedef __attribute__((ext_vector_type(16))) __bf16 v16bf;
typedef __attribute__((ext_vector_type(8)))  float  v8f;
typedef unsigned short ushort_t;

#define BN    4096
#define DIM   2048
#define NHEAD 8
#define HS    256
#define PDIM  2730
#define PPAD  2752             // PDIM padded to multiple of 64
#define NT_UP 172              // up-proj N-tiles of 16 (172*16 = 2752)
#define NT_DN 128              // down-proj N-tiles (128*16 = 2048)
static const size_t NB = (size_t)BN * DIM;   // 8388608

// ---- float -> bf16 (round to nearest even) --------------------------------
__device__ __forceinline__ ushort_t f2bf(float f) {
  unsigned int u = __float_as_uint(f);
  unsigned int r = u + 0x7fffu + ((u >> 16) & 1u);
  return (ushort_t)(r >> 16);
}

// ---- block reduction ------------------------------------------------------
__device__ __forceinline__ float block_reduce_sum(float v, float* sbuf) {
  int tid = threadIdx.x;
  sbuf[tid] = v;
  __syncthreads();
  for (int s = blockDim.x >> 1; s > 0; s >>= 1) {
    if (tid < s) sbuf[tid] += sbuf[tid + s];
    __syncthreads();
  }
  float r = sbuf[0];
  __syncthreads();
  return r;
}

// ---------------------------------------------------------------------------
// WMMA fragment loaders (wave32; contiguous k order, not ISA table order; B swizzle matches)
// A fragment 16x32 (MxK): lane L(0..15) row M=L, K {0..15};
//                         lanes 16..31 same rows, K {16..31}
// -> two contiguous 16-byte runs per lane from row-major A.
// ---------------------------------------------------------------------------
__device__ __forceinline__ v16bf load_a_frag(const ushort_t* __restrict__ A,
                                             int lda, int m0, int k0) {
  int lane  = threadIdx.x & 31;
  int row   = m0 + (lane & 15);
  int khalf = (lane >> 4) * 16;
  const ushort_t* p = A + (size_t)row * lda + k0 + khalf;
  union { uint4 q[2]; v16bf f; } u;
  u.q[0] = *(const uint4*)(p);        // K = khalf + 0..7
  u.q[1] = *(const uint4*)(p + 8);    // K = khalf + 8..15
  return u.f;
}

// B fragment from pre-swizzled weights: fragment (kt,nt) is 512 contiguous
// bf16 = 32 lanes x 16 elements; per-lane read = 32 contiguous bytes.
__device__ __forceinline__ v16bf load_b_swz(const ushort_t* __restrict__ Ws,
                                            int Ntpad, int kt, int nt) {
  int lane = threadIdx.x & 31;
  const ushort_t* p = Ws + (((size_t)kt * Ntpad + nt) * 32 + lane) * 16;
  union { uint4 q[2]; v16bf f; } u;
  u.q[0] = *(const uint4*)(p);
  u.q[1] = *(const uint4*)(p + 8);
  return u.f;
}

#define WMMA_BF16(A_, B_, C_) \
  __builtin_amdgcn_wmma_f32_16x16x32_bf16(false, (A_), false, (B_), (short)0, (C_), false, false)

// ---------------------------------------------------------------------------
// Weight swizzle: f32 [K,N] row-major -> bf16 B-fragment order.
// dst elem i: e=i&15, lane=(i>>4)&31, frag=i>>9, nt=frag%Ntpad, kt=frag/Ntpad
//   k = kt*32 + (lane>>4)*16 + e ;  n = nt*16 + (lane&15)
// OOB (k>=K || n>=N) -> 0, so GEMM inner loop needs no guards.
// ---------------------------------------------------------------------------
__global__ void swz_b_kernel(const float* __restrict__ src,
                             ushort_t* __restrict__ dst,
                             int K, int N, int Kpad, int Ntpad) {
  size_t total = (size_t)Kpad * Ntpad * 16;
  for (size_t i = (size_t)blockIdx.x * blockDim.x + threadIdx.x; i < total;
       i += (size_t)gridDim.x * blockDim.x) {
    int e    = (int)(i & 15);
    int lane = (int)((i >> 4) & 31);
    size_t frag = i >> 9;
    int nt = (int)(frag % Ntpad);
    int kt = (int)(frag / Ntpad);
    int k = kt * 32 + (lane >> 4) * 16 + e;
    int n = nt * 16 + (lane & 15);
    ushort_t v = 0;
    if (k < K && n < N) v = f2bf(src[(size_t)k * N + n]);
    dst[i] = v;
  }
}

// Block-diagonal weights [H,256,256] -> per-head swizzled (Ntpad=16, Kpad=256)
__global__ void swz_bd_kernel(const float* __restrict__ src,
                              ushort_t* __restrict__ dst) {
  size_t total = (size_t)NHEAD * HS * HS;
  for (size_t i = (size_t)blockIdx.x * blockDim.x + threadIdx.x; i < total;
       i += (size_t)gridDim.x * blockDim.x) {
    int head = (int)(i >> 16);
    int r    = (int)(i & 65535);
    int e    = r & 15;
    int lane = (r >> 4) & 31;
    int frag = r >> 9;            // 0..127
    int nt = frag & 15;
    int kt = frag >> 4;
    int k = kt * 32 + (lane >> 4) * 16 + e;
    int n = nt * 16 + (lane & 15);
    dst[i] = f2bf(src[(size_t)head * HS * HS + (size_t)k * HS + n]);
  }
}

// ---------------------------------------------------------------------------
// Kernel: LayerNorm + causal conv1d(k=4) + SiLU; also bf16 copy of raw x.
// One block (256 thr) per row of 2048.
// ---------------------------------------------------------------------------
__global__ void ln_conv_kernel(const float* __restrict__ x,
                               const float* __restrict__ ln_w,
                               const float* __restrict__ ln_b,
                               const float* __restrict__ conv_w,
                               const float* __restrict__ conv_b,
                               ushort_t* __restrict__ x_bf,
                               ushort_t* __restrict__ xc_bf) {
  __shared__ float xn[DIM];
  __shared__ float red[256];
  int row = blockIdx.x, tid = threadIdx.x;
  const float* xr = x + (size_t)row * DIM;
  float vals[8];
  float s = 0.f, ss = 0.f;
#pragma unroll
  for (int j = 0; j < 8; ++j) {
    float v = xr[tid + j * 256];
    vals[j] = v; s += v; ss += v * v;
  }
  float sum   = block_reduce_sum(s, red);
  float sumsq = block_reduce_sum(ss, red);
  float mu  = sum * (1.0f / DIM);
  float var = sumsq * (1.0f / DIM) - mu * mu;
  float rs  = rsqrtf(var + 1e-5f);
#pragma unroll
  for (int j = 0; j < 8; ++j) {
    int d = tid + j * 256;
    xn[d] = (vals[j] - mu) * rs * ln_w[d] + ln_b[d];
    x_bf[(size_t)row * DIM + d] = f2bf(vals[j]);
  }
  __syncthreads();
  float w0 = conv_w[0], w1 = conv_w[1], w2 = conv_w[2], w3 = conv_w[3];
  float cb = conv_b[0];
#pragma unroll
  for (int j = 0; j < 8; ++j) {
    int d = tid + j * 256;
    float a = cb + w3 * xn[d];
    if (d >= 1) a += w2 * xn[d - 1];
    if (d >= 2) a += w1 * xn[d - 2];
    if (d >= 3) a += w0 * xn[d - 3];
    float sl = a / (1.0f + expf(-a));           // SiLU
    xc_bf[(size_t)row * DIM + d] = f2bf(sl);
  }
}

// ---- generic f32 -> bf16 --------------------------------------------------
__global__ void cvt_bf16_kernel(const float* __restrict__ src,
                                ushort_t* __restrict__ dst, size_t n) {
  for (size_t i = (size_t)blockIdx.x * blockDim.x + threadIdx.x; i < n;
       i += (size_t)gridDim.x * blockDim.x)
    dst[i] = f2bf(src[i]);
}

// ---------------------------------------------------------------------------
// Block-diagonal gate GEMM: out[:, h*256+n] = A1@W1[h] + A2@W2[h] + b1 + b2
// Wave tile 32M x 64N (8 accumulators); block = 128 thr = 4 waves (128M).
// Grid: (4 n-tiles, 32 m-tiles, 8 heads). W1/W2 pre-swizzled per head.
// ---------------------------------------------------------------------------
__global__ void bd_gemm_kernel(const ushort_t* __restrict__ A1,
                               const ushort_t* __restrict__ W1,
                               const float* __restrict__ b1,
                               const ushort_t* __restrict__ A2,
                               const ushort_t* __restrict__ W2,
                               const float* __restrict__ b2,
                               float* __restrict__ out) {
  int head = blockIdx.z;
  int wave = threadIdx.x >> 5;
  int m0  = blockIdx.y * 128 + wave * 32;
  int n0  = blockIdx.x * 64;                    // within-head column base
  int ntb = blockIdx.x * 4;
  const ushort_t* W1h = W1 + (size_t)head * HS * HS;
  const ushort_t* W2h = W2 + (size_t)head * HS * HS;

  v8f acc[8];
  v8f zero = {0, 0, 0, 0, 0, 0, 0, 0};
#pragma unroll
  for (int t = 0; t < 8; ++t) acc[t] = zero;

  for (int kt = 0; kt < HS / 32; ++kt) {
    int k0 = kt * 32;
    v16bf a0 = load_a_frag(A1, DIM, m0,      head * HS + k0);
    v16bf a1 = load_a_frag(A1, DIM, m0 + 16, head * HS + k0);
#pragma unroll
    for (int t = 0; t < 4; ++t) {
      v16bf b = load_b_swz(W1h, 16, kt, ntb + t);
      acc[t]     = WMMA_BF16(a0, b, acc[t]);
      acc[4 + t] = WMMA_BF16(a1, b, acc[4 + t]);
    }
    v16bf c0 = load_a_frag(A2, DIM, m0,      head * HS + k0);
    v16bf c1 = load_a_frag(A2, DIM, m0 + 16, head * HS + k0);
#pragma unroll
    for (int t = 0; t < 4; ++t) {
      v16bf b = load_b_swz(W2h, 16, kt, ntb + t);
      acc[t]     = WMMA_BF16(c0, b, acc[t]);
      acc[4 + t] = WMMA_BF16(c1, b, acc[4 + t]);
    }
  }

  int lane = threadIdx.x & 31;
  int colb = lane & 15;
  int rowoff = (lane >> 4) * 8;
#pragma unroll
  for (int half = 0; half < 2; ++half) {
#pragma unroll
    for (int t = 0; t < 4; ++t) {
      int nl = n0 + t * 16 + colb;
      float bv = b1[head * HS + nl] + b2[head * HS + nl];
#pragma unroll
      for (int r = 0; r < 8; ++r) {
        int m = m0 + half * 16 + rowoff + r;
        out[(size_t)m * DIM + head * HS + nl] = acc[half * 4 + t][r] + bv;
      }
    }
  }
}

// ---------------------------------------------------------------------------
// Generic bf16 GEMM: out = A[M,K]@W + bias (+ residual). M=4096.
// Wave tile 32M x 64N; block 128 thr (128M); grid (Ntpad/4, 32).
// ---------------------------------------------------------------------------
__global__ void gemm_bf16_kernel(const ushort_t* __restrict__ A, int lda,
                                 const ushort_t* __restrict__ Ws, int Ntpad,
                                 int Kiter, int N,
                                 const float* __restrict__ bias,
                                 const float* __restrict__ residual, int ldres,
                                 float* __restrict__ out, int ldout) {
  int wave = threadIdx.x >> 5;
  int m0  = blockIdx.y * 128 + wave * 32;
  int n0  = blockIdx.x * 64;
  int ntb = blockIdx.x * 4;

  v8f acc[8];
  v8f zero = {0, 0, 0, 0, 0, 0, 0, 0};
#pragma unroll
  for (int t = 0; t < 8; ++t) acc[t] = zero;

  int nkt = Kiter >> 5;
  for (int kt = 0; kt < nkt; ++kt) {
    int k0 = kt * 32;
    v16bf a0 = load_a_frag(A, lda, m0,      k0);
    v16bf a1 = load_a_frag(A, lda, m0 + 16, k0);
#pragma unroll
    for (int t = 0; t < 4; ++t) {
      v16bf b = load_b_swz(Ws, Ntpad, kt, ntb + t);
      acc[t]     = WMMA_BF16(a0, b, acc[t]);
      acc[4 + t] = WMMA_BF16(a1, b, acc[4 + t]);
    }
  }

  int lane = threadIdx.x & 31;
  int colb = lane & 15;
  int rowoff = (lane >> 4) * 8;
#pragma unroll
  for (int half = 0; half < 2; ++half) {
#pragma unroll
    for (int t = 0; t < 4; ++t) {
      int n = n0 + t * 16 + colb;
      if (n < N) {
        float bv = bias[n];
#pragma unroll
        for (int r = 0; r < 8; ++r) {
          int m = m0 + half * 16 + rowoff + r;
          float v = acc[half * 4 + t][r] + bv;
          if (residual) v += residual[(size_t)m * ldres + n];
          out[(size_t)m * ldout + n] = v;
        }
      }
    }
  }
}

// ---------------------------------------------------------------------------
// Gate math (stabilized exponential gating), writes h/c/n/m into d_out slices.
// ---------------------------------------------------------------------------
__global__ void gate_kernel(const float* __restrict__ g,   // [4][NB] z,i,f,o
                            const float* __restrict__ c_prev,
                            const float* __restrict__ n_prev,
                            const float* __restrict__ m_prev,
                            float* __restrict__ out) {      // d_out base
  size_t nb = NB;
  for (size_t i = (size_t)blockIdx.x * blockDim.x + threadIdx.x; i < nb;
       i += (size_t)gridDim.x * blockDim.x) {
    float zp = g[i];
    float ip = g[nb + i];
    float fp = g[2 * nb + i];
    float op = g[3 * nb + i];
    float mp = m_prev[i];
    float z  = tanhf(zp);
    float o  = 1.0f / (1.0f + expf(-op));
    float mt = fmaxf(fp + mp, ip);
    float ig = expf(ip - mt);
    float fg = expf(fp + mp - mt);
    float ct = fg * c_prev[i] + ig * z;
    float nt = fg * n_prev[i] + ig;
    float ht = o * ct / nt;
    out[nb + i]     = ht;   // h_t
    out[2 * nb + i] = ct;   // c_t
    out[3 * nb + i] = nt;   // n_t
    out[4 * nb + i] = mt;   // m_t
  }
}

// ---------------------------------------------------------------------------
// GroupNorm over each (row, head) group of 256 -> bf16 for the MLP GEMMs.
// ---------------------------------------------------------------------------
__global__ void groupnorm_kernel(const float* __restrict__ h,
                                 const float* __restrict__ gn_w,
                                 const float* __restrict__ gn_b,
                                 ushort_t* __restrict__ gn_bf) {
  __shared__ float red[256];
  int grp = blockIdx.x;
  int b = grp >> 3, head = grp & 7;
  int tid = threadIdx.x;
  size_t base = (size_t)b * DIM + head * HS;
  float v = h[base + tid];
  float s  = block_reduce_sum(v, red);
  float ss = block_reduce_sum(v * v, red);
  float mu  = s * (1.0f / HS);
  float var = ss * (1.0f / HS) - mu * mu;
  float rs  = rsqrtf(var + 1e-5f);
  int d = head * HS + tid;
  float o = (v - mu) * rs * gn_w[d] + gn_b[d];
  gn_bf[base + tid] = f2bf(o);
}

// ---------------------------------------------------------------------------
// act = left * gelu_exact(right), packed bf16 with K padded to 2752.
// ---------------------------------------------------------------------------
__global__ void act_kernel(const float* __restrict__ left,
                           const float* __restrict__ right,
                           ushort_t* __restrict__ act) {
  size_t n = (size_t)BN * PPAD;
  for (size_t i = (size_t)blockIdx.x * blockDim.x + threadIdx.x; i < n;
       i += (size_t)gridDim.x * blockDim.x) {
    int col = (int)(i % PPAD);
    int m   = (int)(i / PPAD);
    ushort_t v = 0;
    if (col < PDIM) {
      float l = left[(size_t)m * PDIM + col];
      float r = right[(size_t)m * PDIM + col];
      float gel = 0.5f * r * (1.0f + erff(r * 0.70710678118654752f));
      v = f2bf(l * gel);
    }
    act[i] = v;
  }
}

// ---------------------------------------------------------------------------
// Host-side launch
// ---------------------------------------------------------------------------
extern "C" void kernel_launch(void* const* d_in, const int* in_sizes, int n_in,
                              void* d_out, int out_size, void* d_ws, size_t ws_size,
                              hipStream_t stream) {
  (void)in_sizes; (void)n_in; (void)out_size; (void)ws_size;

  const float* x      = (const float*)d_in[0];
  const float* h_prev = (const float*)d_in[1];
  const float* c_prev = (const float*)d_in[2];
  const float* n_prev = (const float*)d_in[3];
  const float* m_prev = (const float*)d_in[4];
  const float* ln_w   = (const float*)d_in[5];
  const float* ln_b   = (const float*)d_in[6];
  const float* conv_w = (const float*)d_in[7];
  const float* conv_b = (const float*)d_in[8];
  const float* gn_w   = (const float*)d_in[9];
  const float* gn_b   = (const float*)d_in[10];
  const float* upLw   = (const float*)d_in[11];
  const float* upLb   = (const float*)d_in[12];
  const float* upRw   = (const float*)d_in[13];
  const float* upRb   = (const float*)d_in[14];
  const float* downw  = (const float*)d_in[15];
  const float* downb  = (const float*)d_in[16];
  // Wz 17/18 Wi 19/20 Wf 21/22 Wo 23/24 Rz 25/26 Ri 27/28 Rf 29/30 Ro 31/32
  const float* bias_g[8];
  const int wIdx[8] = {17, 19, 21, 23, 25, 27, 29, 31};
  for (int g = 0; g < 8; ++g) bias_g[g] = (const float*)d_in[wIdx[g] + 1];

  float* out = (float*)d_out;

  // ---- workspace layout ----
  const size_t WBLK_E = (size_t)NHEAD * HS * HS;        // 524288 per gate
  const size_t UPW_E  = (size_t)DIM * NT_UP * 16;       // 2048*2752 swizzled
  const size_t DNW_E  = (size_t)PPAD * NT_DN * 16;      // 2752*2048 swizzled
  char* ws = (char*)d_ws;
  size_t off = 0;
  float*    gates  = (float*)(ws + off);    off += 4 * NB * sizeof(float); // reused for left/right
  ushort_t* xbf    = (ushort_t*)(ws + off); off += NB * 2;
  ushort_t* xcbf   = (ushort_t*)(ws + off); off += NB * 2;
  ushort_t* hbf    = (ushort_t*)(ws + off); off += NB * 2;
  ushort_t* gnbf   = (ushort_t*)(ws + off); off += NB * 2;
  ushort_t* actbf  = (ushort_t*)(ws + off); off += (size_t)BN * PPAD * 2;
  ushort_t* wblk   = (ushort_t*)(ws + off); off += 8 * WBLK_E * 2;
  ushort_t* upLbf  = (ushort_t*)(ws + off); off += UPW_E * 2;
  ushort_t* upRbf  = (ushort_t*)(ws + off); off += UPW_E * 2;
  ushort_t* downbf = (ushort_t*)(ws + off); off += DNW_E * 2;

  // ---- stage operands (bf16 + B-fragment swizzle for all weights) ----
  cvt_bf16_kernel<<<2048, 256, 0, stream>>>(h_prev, hbf, NB);
  for (int g = 0; g < 8; ++g)
    swz_bd_kernel<<<512, 256, 0, stream>>>((const float*)d_in[wIdx[g]],
                                           wblk + (size_t)g * WBLK_E);
  swz_b_kernel<<<1024, 256, 0, stream>>>(upLw, upLbf, DIM, PDIM, DIM, NT_UP);
  swz_b_kernel<<<1024, 256, 0, stream>>>(upRw, upRbf, DIM, PDIM, DIM, NT_UP);
  swz_b_kernel<<<1024, 256, 0, stream>>>(downw, downbf, PDIM, DIM, PPAD, NT_DN);

  // ---- LN + causal conv + SiLU (+ raw x bf16) ----
  ln_conv_kernel<<<BN, 256, 0, stream>>>(x, ln_w, ln_b, conv_w, conv_b, xbf, xcbf);

  // ---- block-diagonal gate pre-activations ----
  dim3 bdGrid(4, 32, NHEAD);
  bd_gemm_kernel<<<bdGrid, 128, 0, stream>>>(xbf,  wblk + 0 * WBLK_E, bias_g[0],
                                             hbf,  wblk + 4 * WBLK_E, bias_g[4],
                                             gates + 0 * NB);   // z
  bd_gemm_kernel<<<bdGrid, 128, 0, stream>>>(xcbf, wblk + 1 * WBLK_E, bias_g[1],
                                             hbf,  wblk + 5 * WBLK_E, bias_g[5],
                                             gates + 1 * NB);   // i
  bd_gemm_kernel<<<bdGrid, 128, 0, stream>>>(xcbf, wblk + 2 * WBLK_E, bias_g[2],
                                             hbf,  wblk + 6 * WBLK_E, bias_g[6],
                                             gates + 2 * NB);   // f
  bd_gemm_kernel<<<bdGrid, 128, 0, stream>>>(xbf,  wblk + 3 * WBLK_E, bias_g[3],
                                             hbf,  wblk + 7 * WBLK_E, bias_g[7],
                                             gates + 3 * NB);   // o

  // ---- gates -> h_t/c_t/n_t/m_t (into d_out slices) ----
  gate_kernel<<<4096, 256, 0, stream>>>(gates, c_prev, n_prev, m_prev, out);

  // ---- GroupNorm(h_t) -> bf16 ----
  groupnorm_kernel<<<BN * NHEAD, 256, 0, stream>>>(out + NB, gn_w, gn_b, gnbf);

  // ---- MLP: up-projections (reuse gate buffer), GELU gate, down + residual ----
  float* left  = gates;                 // [4096, 2730] f32
  float* right = gates + 2 * NB;        // [4096, 2730] f32
  dim3 upGrid(NT_UP / 4, 32);           // 43 x 32
  gemm_bf16_kernel<<<upGrid, 128, 0, stream>>>(gnbf, DIM, upLbf, NT_UP, DIM, PDIM,
                                               upLb, nullptr, 0, left, PDIM);
  gemm_bf16_kernel<<<upGrid, 128, 0, stream>>>(gnbf, DIM, upRbf, NT_UP, DIM, PDIM,
                                               upRb, nullptr, 0, right, PDIM);
  act_kernel<<<2048, 256, 0, stream>>>(left, right, actbf);

  dim3 dnGrid(NT_DN / 4, 32);           // 32 x 32
  gemm_bf16_kernel<<<dnGrid, 128, 0, stream>>>(actbf, PPAD, downbf, NT_DN, PPAD, DIM,
                                               downb, x, DIM, out, DIM);
}
